// VerificationModule_11553462026576
// MI455X (gfx1250) — matrix operand order repaired and checked by reading, not compiled, not run on hardware
//
#include <hip/hip_runtime.h>
#include <hip/hip_bf16.h>

typedef __attribute__((ext_vector_type(16))) _Float16 v16h;
typedef __attribute__((ext_vector_type(8)))  _Float16 v8h;
typedef __attribute__((ext_vector_type(8)))  float    v8f;

// ---------------------------------------------------------------------------
// Helpers
// ---------------------------------------------------------------------------

__device__ __forceinline__ float gelu_exact(float x) {
    return 0.5f * x * (1.0f + erff(x * 0.70710678118654752f));
}

// A-tile (16x32 f16) from LDS row-major [rows][stride].
// CDNA5 layout: lane L -> row M = L%16; halves i<8: K=(L/16)*8+i ; i>=8: K=16+(L/16)*8+(i-8)
__device__ __forceinline__ v16h load_a_lds(const _Float16* base, int stride,
                                           int r0, int k0, int lane) {
    const _Float16* p = base + (r0 + (lane & 15)) * stride + k0 + ((lane >> 4) << 3);
    v8h lo = *(const v8h*)p;
    v8h hi = *(const v8h*)(p + 16);
    return __builtin_shufflevector(lo, hi, 0,1,2,3,4,5,6,7,8,9,10,11,12,13,14,15);
}

// B-tile (32x16 f16) from global transposed f16 weights W^T [Ncols][Kw] row-major.
// CDNA5 layout: lane L -> col N = L%16; halves i<8: K=(L/16)*8+i ; i>=8: K=16+(L/16)*8+(i-8)
__device__ __forceinline__ v16h load_b_glb(const _Float16* wT, int Kw,
                                           int c0, int k0, int lane) {
    const _Float16* p = wT + (size_t)(c0 + (lane & 15)) * Kw + k0 + ((lane >> 4) << 3);
    v8h lo = *(const v8h*)p;
    v8h hi = *(const v8h*)(p + 16);
    return __builtin_shufflevector(lo, hi, 0,1,2,3,4,5,6,7,8,9,10,11,12,13,14,15);
}

// Register-blocked GEMM: NRT 16-row tiles x NCT 16-col tiles per wave.
// K-loop kept rolled (#pragma unroll 1) so only one k-step of operands is live:
// acc NRT*NCT*8 + A NRT*8 + B NCT*8 VGPRs -> no spills, no high-VGPR MSB traffic.
template <int NRT, int NCT, int KDIM>
__device__ __forceinline__ void mm_block(const _Float16* aLds, const _Float16* wT,
                                         int c0, int lane, v8f (&acc)[NRT][NCT]) {
#pragma unroll
    for (int r = 0; r < NRT; ++r)
#pragma unroll
        for (int c = 0; c < NCT; ++c) acc[r][c] = {};

#pragma unroll 1
    for (int k0 = 0; k0 < KDIM; k0 += 32) {
        v16h b[NCT];
#pragma unroll
        for (int c = 0; c < NCT; ++c) b[c] = load_b_glb(wT, KDIM, c0 + (c << 4), k0, lane);
        v16h a[NRT];
#pragma unroll
        for (int r = 0; r < NRT; ++r) a[r] = load_a_lds(aLds, KDIM, r << 4, k0, lane);
#pragma unroll
        for (int r = 0; r < NRT; ++r)
#pragma unroll
            for (int c = 0; c < NCT; ++c)
                acc[r][c] = __builtin_amdgcn_wmma_f32_16x16x32_f16(
                                false, a[r], false, b[c], (short)0, acc[r][c], false, false);
    }
}

// ---------------------------------------------------------------------------
// Kernel 0: transpose + f32->f16 convert weights: src [K][Nc] -> dst [Nc][K]
// ---------------------------------------------------------------------------
__global__ void transpose_cvt_f16(const float* __restrict__ src,
                                  _Float16* __restrict__ dst, int K, int Nc) {
    int i = blockIdx.x * 256 + threadIdx.x;
    if (i < K * Nc) {
        int k = i / Nc;
        int n = i - k * Nc;
        dst[(size_t)n * K + k] = (_Float16)src[i];
    }
}

// ---------------------------------------------------------------------------
// One tree level, x-rows = NRT*16. GEMM A (K=512, gelu) then GEMM B (K=256).
// Each wave owns a 32-column strip (NCT=2). NRT==1 is the final level:
// emit rule partial sums instead of eBuf rows.
// ---------------------------------------------------------------------------
template <int NRT>
__device__ __forceinline__ void tree_level(
    _Float16* eBuf, _Float16* hBuf,
    const _Float16* __restrict__ WcT1, const float* __restrict__ bc1,
    const _Float16* __restrict__ WcT2, const float* __restrict__ bc2,
    int lane, int wave, int blockId, float* __restrict__ ruleBuf) {

    const int c0 = wave << 5;              // 8 waves x 32 cols = 256 cols
    const int rb = (lane >> 4) << 3;

    // GEMM A: x[NRT*16][512] @ Wc1 -> gelu -> hBuf[NRT*16][256]
    {
        v8f acc[NRT][2];
        mm_block<NRT, 2, 512>(eBuf, WcT1, c0, lane, acc);
#pragma unroll
        for (int c = 0; c < 2; ++c) {
            int   col  = c0 + (c << 4) + (lane & 15);
            float bias = bc1[col];
#pragma unroll
            for (int r = 0; r < NRT; ++r)
#pragma unroll
                for (int v = 0; v < 8; ++v)
                    hBuf[((r << 4) + rb + v) * 256 + col] =
                        (_Float16)gelu_exact(acc[r][c][v] + bias);
        }
    }
    __syncthreads();

    // GEMM B: hBuf[NRT*16][256] @ Wc2 (+bc2) -> eBuf (or rule at final level)
    {
        v8f acc[NRT][2];
        mm_block<NRT, 2, 256>(hBuf, WcT2, c0, lane, acc);
#pragma unroll
        for (int c = 0; c < 2; ++c) {
            int   col  = c0 + (c << 4) + (lane & 15);
            float bias = bc2[col];
            if constexpr (NRT > 1) {
#pragma unroll
                for (int r = 0; r < NRT; ++r)
#pragma unroll
                    for (int v = 0; v < 8; ++v)
                        eBuf[((r << 4) + rb + v) * 256 + col] =
                            (_Float16)(acc[r][c][v] + bias);
            } else {
                // Final level: lanes 0-15 hold rows 0..7 (= n0), lanes 16-31 rows 8..15 (= n1).
                // rule[n][col] = (1/16) * sum of the 8 args of n (pred term added in head kernel)
                float s = 0.0f;
#pragma unroll
                for (int v = 0; v < 8; ++v) s += acc[0][c][v];
                s += 8.0f * bias;
                int n = (blockId << 1) + (lane >> 4);
                ruleBuf[(size_t)n * 256 + col] = s * 0.0625f;
            }
        }
    }
    __syncthreads();
}

// ---------------------------------------------------------------------------
// Kernel 1: leaf gather + 3-level tree reduction + rule accumulation.
// One workgroup = 16 statements = 2 complete n-rows (A*R = 8 statements per n).
// ---------------------------------------------------------------------------
__global__ __launch_bounds__(256) void tree_kernel(
    const float* __restrict__ const_emb, const float* __restrict__ var_emb,
    const _Float16* __restrict__ WcT1, const float* __restrict__ bc1,
    const _Float16* __restrict__ WcT2, const float* __restrict__ bc2,
    const int* __restrict__ leaf_idx, const int* __restrict__ leaf_type,
    float* __restrict__ ruleBuf) {

    __shared__ _Float16 eBuf[128 * 256];   // 64 KB: leaf/intermediate embeddings (f16)
    __shared__ _Float16 hBuf[64 * 256];    // 32 KB: gelu hidden (f16)

    const int tid  = threadIdx.x;
    const int lane = tid & 31;
    const int wave = tid >> 5;
    const int s0   = blockIdx.x << 4;      // first statement of this WG

    // ---- Gather: 128 leaf rows of 256 f32 -> f16 into eBuf ----
    for (int u = tid; u < 128 * 64; u += 256) {
        int row = u >> 6;                  // 0..127 leaf row
        int c4  = u & 63;                  // float4 chunk within row
        int g   = (s0 << 3) + row;         // flat leaf index
        int idx = leaf_idx[g];
        int typ = leaf_type[g];
        const float* src = (typ == 1) ? (var_emb + (size_t)(idx & 255) * 256)
                                      : (const_emb + (size_t)idx * 256);
        float4 f = *(const float4*)(src + (c4 << 2));
        _Float16* dst = &eBuf[row * 256 + (c4 << 2)];
        dst[0] = (_Float16)f.x; dst[1] = (_Float16)f.y;
        dst[2] = (_Float16)f.z; dst[3] = (_Float16)f.w;
    }
    __syncthreads();

    // ---- Tree levels: x-rows 64 -> 32 -> 16 ----
    tree_level<4>(eBuf, hBuf, WcT1, bc1, WcT2, bc2, lane, wave, blockIdx.x, ruleBuf);
    tree_level<2>(eBuf, hBuf, WcT1, bc1, WcT2, bc2, lane, wave, blockIdx.x, ruleBuf);
    tree_level<1>(eBuf, hBuf, WcT1, bc1, WcT2, bc2, lane, wave, blockIdx.x, ruleBuf);
}

// ---------------------------------------------------------------------------
// Kernel 2: head MLP. 16 n-rows per workgroup.
// rule_full = ruleBuf + (1/8) * sum_a const_emb[pred_idx[n,a]]
// scores = sigmoid(gelu(gelu(rule@Wv1+bv1)@Wv2+bv2)@Wv3+bv3)
// ---------------------------------------------------------------------------
__global__ __launch_bounds__(256) void head_kernel(
    const float* __restrict__ const_emb,
    const _Float16* __restrict__ WvT1, const float* __restrict__ bv1,
    const _Float16* __restrict__ WvT2, const float* __restrict__ bv2,
    const float* __restrict__ Wv3, const float* __restrict__ bv3,
    const int* __restrict__ pred_idx,
    const float* __restrict__ ruleBuf,
    float* __restrict__ out) {

    __shared__ _Float16 ruleLds[16 * 256];
    __shared__ _Float16 h1Lds[16 * 256];
    __shared__ float    h2Lds[16 * 128];

    const int tid  = threadIdx.x;
    const int lane = tid & 31;
    const int wave = tid >> 5;
    const int n0   = blockIdx.x << 4;

    // Stage rule (+ pred embedding mean term) as f16
    for (int i = tid; i < 16 * 256; i += 256) {
        int row = i >> 8;
        int d   = i & 255;
        int n   = n0 + row;
        const int* pp = pred_idx + n * 4;
        float v = ruleBuf[(size_t)n * 256 + d] +
                  0.125f * (const_emb[(size_t)pp[0] * 256 + d] +
                            const_emb[(size_t)pp[1] * 256 + d] +
                            const_emb[(size_t)pp[2] * 256 + d] +
                            const_emb[(size_t)pp[3] * 256 + d]);
        ruleLds[i] = (_Float16)v;
    }
    __syncthreads();

    // Stage 1: [16][256] @ Wv1 -> gelu -> h1 (one 32-col strip per wave)
    {
        int c0 = wave << 5;
        v8f acc[1][2];
        mm_block<1, 2, 256>(ruleLds, WvT1, c0, lane, acc);
        int rb = (lane >> 4) << 3;
#pragma unroll
        for (int c = 0; c < 2; ++c) {
            int   col  = c0 + (c << 4) + (lane & 15);
            float bias = bv1[col];
#pragma unroll
            for (int v = 0; v < 8; ++v)
                h1Lds[(rb + v) * 256 + col] = (_Float16)gelu_exact(acc[0][c][v] + bias);
        }
    }
    __syncthreads();

    // Stage 2: [16][256] @ Wv2 -> gelu -> h2 f32 (one 16-col tile per wave)
    {
        int c0 = wave << 4;
        v8f acc[1][1];
        mm_block<1, 1, 256>(h1Lds, WvT2, c0, lane, acc);
        int   col  = c0 + (lane & 15);
        float bias = bv2[col];
        int   rb   = (lane >> 4) << 3;
#pragma unroll
        for (int v = 0; v < 8; ++v)
            h2Lds[(rb + v) * 128 + col] = gelu_exact(acc[0][0][v] + bias);
    }
    __syncthreads();

    // Final 128-dot + sigmoid
    if (tid < 16) {
        float s = bv3[0];
        for (int k = 0; k < 128; ++k) s += h2Lds[tid * 128 + k] * Wv3[k];
        out[n0 + tid] = 1.0f / (1.0f + expf(-s));
    }
}

// ---------------------------------------------------------------------------
// Launch
// ---------------------------------------------------------------------------
extern "C" void kernel_launch(void* const* d_in, const int* in_sizes, int n_in,
                              void* d_out, int out_size, void* d_ws, size_t ws_size,
                              hipStream_t stream) {
    const float* const_emb = (const float*)d_in[0];
    const float* var_emb   = (const float*)d_in[1];
    const float* Wc1       = (const float*)d_in[2];
    const float* bc1       = (const float*)d_in[3];
    const float* Wc2       = (const float*)d_in[4];
    const float* bc2       = (const float*)d_in[5];
    const float* Wv1       = (const float*)d_in[6];
    const float* bv1       = (const float*)d_in[7];
    const float* Wv2       = (const float*)d_in[8];
    const float* bv2       = (const float*)d_in[9];
    const float* Wv3       = (const float*)d_in[10];
    const float* bv3       = (const float*)d_in[11];
    const int*   leaf_idx  = (const int*)d_in[12];
    const int*   leaf_type = (const int*)d_in[13];
    const int*   pred_idx  = (const int*)d_in[14];

    char* ws = (char*)d_ws;
    _Float16* WcT1 = (_Float16*)(ws + 0);         // 512*256*2 = 262144 B
    _Float16* WcT2 = (_Float16*)(ws + 262144);    // 256*256*2 = 131072 B
    _Float16* WvT1 = (_Float16*)(ws + 393216);    // 256*256*2 = 131072 B
    _Float16* WvT2 = (_Float16*)(ws + 524288);    // 256*128*2 =  65536 B
    float*    ruleBuf = (float*)(ws + 1048576);   // 4096*256*4 = 4 MB

    transpose_cvt_f16<<<(512 * 256 + 255) / 256, 256, 0, stream>>>(Wc1, WcT1, 512, 256);
    transpose_cvt_f16<<<(256 * 256 + 255) / 256, 256, 0, stream>>>(Wc2, WcT2, 256, 256);
    transpose_cvt_f16<<<(256 * 256 + 255) / 256, 256, 0, stream>>>(Wv1, WvT1, 256, 256);
    transpose_cvt_f16<<<(256 * 128 + 255) / 256, 256, 0, stream>>>(Wv2, WvT2, 256, 128);

    tree_kernel<<<2048, 256, 0, stream>>>(const_emb, var_emb, WcT1, bc1, WcT2, bc2,
                                          leaf_idx, leaf_type, ruleBuf);
    head_kernel<<<256, 256, 0, stream>>>(const_emb, WvT1, bv1, WvT2, bv2, Wv3, bv3,
                                         pred_idx, ruleBuf, (float*)d_out);
}
